// BidirectionalMessagePassing_23502061043722
// MI455X (gfx1250) — compile-verified, matrix operand order checked by reading, not run on hardware
//
#include <hip/hip_runtime.h>
#include <hip/hip_bf16.h>

#define N_IN   128
#define C_DIR  64
#define N_OUT  128
#define NEG    -1e9f
#define BN_EPS 1e-5f

#define ROWS_PER_BLK 64
#define ASTRIDE 136   // f16 elements; 272 B: 16B-aligned rows, banks spread by 4
#define BSTRIDE 136

typedef __attribute__((ext_vector_type(16))) _Float16 v16h;
typedef __attribute__((ext_vector_type(8)))  _Float16 v8h;
typedef __attribute__((ext_vector_type(8)))  float    v8f;

// ---------------------------------------------------------------------------
// Load a v16h fragment whose two 8-element K-groups are contiguous in LDS
// at p and p+16 (elements). Lowers to 2x ds_load_b128.
// ---------------------------------------------------------------------------
__device__ inline v16h frag_ld(const _Float16* p) {
    v8h lo = *(const v8h*)(p);
    v8h hi = *(const v8h*)(p + 16);
    return __builtin_shufflevector(lo, hi, 0, 1, 2, 3, 4, 5, 6, 7,
                                           8, 9, 10, 11, 12, 13, 14, 15);
}
// CDNA5 16x16x32 f16 fragment indexing (wave32):
//   A: lane L holds row M=L%16; element e -> K = (e/8)*16 + (L/16)*8 + (e%8)
//   B: lane L holds col N=n0+L%16; same K indexing.
// With row-major A [m][k] and col-major B [n][k] in LDS, each 8-element group
// is contiguous, so frag_ld() applies to both.

// ---------------------------------------------------------------------------
// Kernel 1: h = x @ [Wf | Wb]   ([N,128] = [N,128] @ [128,128])
// 64 rows per block; 8 waves; wave w owns cols [16w,16w+16) and loops 4 row
// tiles, holding its 4 B fragments (K=128) in registers. 16 WMMA per wave.
// ---------------------------------------------------------------------------
__global__ void gemm_h_kernel(const float* __restrict__ x,
                              const float* __restrict__ Wf,
                              const float* __restrict__ Wb,
                              float* __restrict__ h, int n) {
    __shared__ _Float16 Ash[ROWS_PER_BLK * ASTRIDE];   // ~17 KB, row-major [m][k]
    __shared__ _Float16 Bsh[N_OUT * BSTRIDE];          // ~34 KB, col-major [n][k]
    const int tid = threadIdx.x;
    const int row0 = blockIdx.x * ROWS_PER_BLK;
    const bool full = (row0 + ROWS_PER_BLK) <= n;      // uniform: only last block partial

    // stage x tile (f32 -> f16), coalesced reads, conflict-free writes
    for (int i = tid; i < ROWS_PER_BLK * N_IN; i += 256) {
        int r = i >> 7, k = i & 127;
        int gr = row0 + r;
        Ash[r * ASTRIDE + k] = (full || gr < n) ? (_Float16)x[(size_t)gr * N_IN + k]
                                                : (_Float16)0.0f;
    }
    // stage weights col-major: Bsh[c][k] = (c<64 ? Wf : Wb)[k][c%64]
    for (int i = tid; i < N_IN * N_OUT; i += 256) {
        int c = i >> 7, k = i & 127;
        float w = (c < C_DIR) ? Wf[k * C_DIR + c] : Wb[k * C_DIR + (c - C_DIR)];
        Bsh[c * BSTRIDE + k] = (_Float16)w;
    }
    __syncthreads();

    const int wave = tid >> 5, lane = tid & 31;
    const int n0 = wave * 16;
    const int half = lane >> 4;
    const int col = n0 + (lane & 15);

    // B fragments for all 4 K-steps, kept live across row tiles
    v16h bfrag[4];
#pragma unroll
    for (int ks = 0; ks < 4; ++ks)
        bfrag[ks] = frag_ld(Bsh + col * BSTRIDE + ks * 32 + half * 8);

#pragma unroll
    for (int rt = 0; rt < 4; ++rt) {
        const _Float16* abase = Ash + (rt * 16 + (lane & 15)) * ASTRIDE + half * 8;
        v8f acc = {};
#pragma unroll
        for (int ks = 0; ks < 4; ++ks) {
            v16h a = frag_ld(abase + ks * 32);
            acc = __builtin_amdgcn_wmma_f32_16x16x32_f16(false, a, false, bfrag[ks],
                                                         (short)0, acc, false, false);
        }
        const int mbase = row0 + rt * 16 + half * 8;
        float* p = h + mbase * N_OUT + col;            // 32-bit index math
        if (full) {
#pragma unroll
            for (int r = 0; r < 8; ++r) p[r * N_OUT] = acc[r];   // imm-offset stores
        } else {
#pragma unroll
            for (int r = 0; r < 8; ++r)
                if (mbase + r < n) p[r * N_OUT] = acc[r];
        }
    }
}

// ---------------------------------------------------------------------------
// Kernel 2: per-node attention scalars (4 dots over 64 channels)
// ---------------------------------------------------------------------------
__global__ void node_alpha_kernel(const float* __restrict__ h,
                                  const float* __restrict__ a_src_f,
                                  const float* __restrict__ a_dst_f,
                                  const float* __restrict__ a_src_b,
                                  const float* __restrict__ a_dst_b,
                                  float* __restrict__ asf, float* __restrict__ adf,
                                  float* __restrict__ asb, float* __restrict__ adb,
                                  int n) {
    int i = blockIdx.x * blockDim.x + threadIdx.x;
    if (i >= n) return;
    float sf = 0.f, df = 0.f, sb = 0.f, db = 0.f;
    const float* hr = h + (size_t)i * N_OUT;
#pragma unroll 8
    for (int c = 0; c < C_DIR; ++c) {
        float vf = hr[c], vb = hr[C_DIR + c];
        sf += vf * a_src_f[c]; df += vf * a_dst_f[c];
        sb += vb * a_src_b[c]; db += vb * a_dst_b[c];
    }
    asf[i] = sf; adf[i] = df; asb[i] = sb; adb[i] = db;
}

// ---------------------------------------------------------------------------
// Kernel 3: 16 per-edge-type scalars: esc[t] = We_f[t,:]@a_e_f ; esc[8+t] = bwd
// ---------------------------------------------------------------------------
__global__ void escore_kernel(const float* __restrict__ We_f,
                              const float* __restrict__ a_e_f,
                              const float* __restrict__ We_b,
                              const float* __restrict__ a_e_b,
                              float* __restrict__ esc) {
    int t = threadIdx.x;
    if (t >= 16) return;
    const float* W = (t < 8) ? We_f + t * C_DIR : We_b + (t - 8) * C_DIR;
    const float* a = (t < 8) ? a_e_f : a_e_b;
    float s = 0.f;
    for (int c = 0; c < C_DIR; ++c) s += W[c] * a[c];
    esc[t] = s;
}

// ---------------------------------------------------------------------------
// Kernel 4: init accumulators: comb seeded with biases; maxes = NEG; denom = 0
// ---------------------------------------------------------------------------
__global__ void init_kernel(float* __restrict__ comb,
                            const float* __restrict__ bf,
                            const float* __restrict__ bb,
                            float* __restrict__ m_f, float* __restrict__ m_b,
                            float* __restrict__ den_f, float* __restrict__ den_b,
                            int n) {
    int i = blockIdx.x * blockDim.x + threadIdx.x;
    int total = n * N_OUT;
    if (i < total) {
        int c = i & 127;
        comb[i] = (c < C_DIR) ? bf[c] : bb[c - C_DIR];
    }
    if (i < n) {
        m_f[i] = NEG; m_b[i] = NEG;
        den_f[i] = 0.f; den_b[i] = 0.f;
    }
}

__device__ inline void atomicMaxFloat(float* addr, float val) {
    if (val >= 0.0f) atomicMax((int*)addr, __float_as_int(val));
    else             atomicMin((unsigned int*)addr, __float_as_uint(val));
}

// ---------------------------------------------------------------------------
// Kernel 5: per-edge logit + segment max (fp32 atomics through L2)
// ---------------------------------------------------------------------------
__global__ void edge_pass1_kernel(const int* __restrict__ src,
                                  const int* __restrict__ dst,
                                  const float* __restrict__ edge_attr,
                                  const float* __restrict__ asf, const float* __restrict__ adf,
                                  const float* __restrict__ asb, const float* __restrict__ adb,
                                  const float* __restrict__ esc,
                                  float* __restrict__ logit, int* __restrict__ emask,
                                  float* __restrict__ m_f, float* __restrict__ m_b,
                                  int e) {
    int i = blockIdx.x * blockDim.x + threadIdx.x;
    if (i >= e) return;
    int s = src[i], d = dst[i];
    int t = 0;
    const float* ea = edge_attr + (size_t)i * 8;
#pragma unroll
    for (int j = 0; j < 8; ++j) if (ea[j] > 0.5f) t = j;
    int fwd = (0x15 >> t) & 1;                 // etype in {0,2,4}
    float lg = fwd ? (asf[s] + adf[d] + esc[t])
                   : (asb[s] + adb[d] + esc[8 + t]);
    lg = (lg > 0.f) ? lg : 0.2f * lg;          // leaky_relu(0.2)
    logit[i] = lg;
    emask[i] = fwd;
    atomicMaxFloat(fwd ? (m_f + d) : (m_b + d), lg);
}

// ---------------------------------------------------------------------------
// Kernel 6: exp(logit - m[dst]) + segment sum
// ---------------------------------------------------------------------------
__global__ void edge_pass2_kernel(const int* __restrict__ dst,
                                  const float* __restrict__ logit,
                                  const int* __restrict__ emask,
                                  const float* __restrict__ m_f,
                                  const float* __restrict__ m_b,
                                  float* __restrict__ den_f, float* __restrict__ den_b,
                                  float* __restrict__ exv, int e) {
    int i = blockIdx.x * blockDim.x + threadIdx.x;
    if (i >= e) return;
    int d = dst[i];
    int fwd = emask[i];
    float m = fwd ? m_f[d] : m_b[d];
    m = (m > NEG * 0.5f) ? m : 0.f;
    float ex = __expf(logit[i] - m);
    exv[i] = ex;
    atomicAdd(fwd ? (den_f + d) : (den_b + d), ex);
}

// ---------------------------------------------------------------------------
// Kernel 7: scatter alpha * h[src] into comb (64 threads per edge)
// ---------------------------------------------------------------------------
__global__ void edge_scatter_kernel(const int* __restrict__ src,
                                    const int* __restrict__ dst,
                                    const int* __restrict__ emask,
                                    const float* __restrict__ exv,
                                    const float* __restrict__ den_f,
                                    const float* __restrict__ den_b,
                                    const float* __restrict__ h,
                                    float* __restrict__ comb, int e) {
    int gid = blockIdx.x * blockDim.x + threadIdx.x;
    int i = gid >> 6;            // edge
    int c = gid & 63;            // channel
    if (i >= e) return;
    int s = src[i], d = dst[i];
    int fwd = emask[i];
    float den = fwd ? den_f[d] : den_b[d];
    den = (den > 0.f) ? den : 1.f;
    float alpha = exv[i] / den;
    int off = fwd ? 0 : C_DIR;
    float v = alpha * h[(size_t)s * N_OUT + off + c];
    atomicAdd(comb + (size_t)d * N_OUT + off + c, v);
}

// ---------------------------------------------------------------------------
// Kernel 8: out = comb @ Wc^T + bc   (WMMA; Wc is [OUT,OUT] row-major (o,k),
// which is exactly the col-major-by-output layout Bsh needs: straight copy.)
// ---------------------------------------------------------------------------
__global__ void gemm_combine_kernel(const float* __restrict__ comb,
                                    const float* __restrict__ Wc,
                                    const float* __restrict__ bc,
                                    float* __restrict__ out, int n) {
    __shared__ _Float16 Ash[ROWS_PER_BLK * ASTRIDE];   // [m][k]
    __shared__ _Float16 Bsh[N_OUT * BSTRIDE];          // [o][k]
    const int tid = threadIdx.x;
    const int row0 = blockIdx.x * ROWS_PER_BLK;
    const bool full = (row0 + ROWS_PER_BLK) <= n;

    for (int i = tid; i < ROWS_PER_BLK * N_OUT; i += 256) {
        int r = i >> 7, k = i & 127;
        int gr = row0 + r;
        Ash[r * ASTRIDE + k] = (full || gr < n) ? (_Float16)comb[(size_t)gr * N_OUT + k]
                                                : (_Float16)0.0f;
    }
    for (int i = tid; i < N_OUT * N_OUT; i += 256) {
        int o = i >> 7, k = i & 127;
        Bsh[o * BSTRIDE + k] = (_Float16)Wc[o * N_OUT + k];  // coalesced copy
    }
    __syncthreads();

    const int wave = tid >> 5, lane = tid & 31;
    const int n0 = wave * 16;
    const int half = lane >> 4;
    const int col = n0 + (lane & 15);
    const float bias = bc[col];

    v16h bfrag[4];
#pragma unroll
    for (int ks = 0; ks < 4; ++ks)
        bfrag[ks] = frag_ld(Bsh + col * BSTRIDE + ks * 32 + half * 8);

#pragma unroll
    for (int rt = 0; rt < 4; ++rt) {
        const _Float16* abase = Ash + (rt * 16 + (lane & 15)) * ASTRIDE + half * 8;
        v8f acc = {};
#pragma unroll
        for (int ks = 0; ks < 4; ++ks) {
            v16h a = frag_ld(abase + ks * 32);
            acc = __builtin_amdgcn_wmma_f32_16x16x32_f16(false, a, false, bfrag[ks],
                                                         (short)0, acc, false, false);
        }
        const int mbase = row0 + rt * 16 + half * 8;
        float* p = out + mbase * N_OUT + col;
        if (full) {
#pragma unroll
            for (int r = 0; r < 8; ++r) p[r * N_OUT] = acc[r] + bias;
        } else {
#pragma unroll
            for (int r = 0; r < 8; ++r)
                if (mbase + r < n) p[r * N_OUT] = acc[r] + bias;
        }
    }
}

// ---------------------------------------------------------------------------
// Kernel 9: batch-norm stats per channel -> scale/shift
// ---------------------------------------------------------------------------
__global__ void bn_stats_kernel(const float* __restrict__ out,
                                const float* __restrict__ gamma,
                                const float* __restrict__ beta,
                                float* __restrict__ scale, float* __restrict__ shift,
                                int n) {
    __shared__ float s1[256], s2[256];
    int c = blockIdx.x, tid = threadIdx.x;
    float s = 0.f, sq = 0.f;
    for (int i = tid; i < n; i += 256) {
        float v = out[(size_t)i * N_OUT + c];
        s += v; sq += v * v;
    }
    s1[tid] = s; s2[tid] = sq; __syncthreads();
    for (int o = 128; o > 0; o >>= 1) {
        if (tid < o) { s1[tid] += s1[tid + o]; s2[tid] += s2[tid + o]; }
        __syncthreads();
    }
    if (tid == 0) {
        float mean = s1[0] / (float)n;
        float var = s2[0] / (float)n - mean * mean;
        float sc = gamma[c] * rsqrtf(var + BN_EPS);
        scale[c] = sc;
        shift[c] = beta[c] - mean * sc;
    }
}

// ---------------------------------------------------------------------------
// Kernel 10: in-place BN apply + ReLU
// ---------------------------------------------------------------------------
__global__ void bn_apply_kernel(float* __restrict__ out,
                                const float* __restrict__ scale,
                                const float* __restrict__ shift,
                                int total) {
    int i = blockIdx.x * blockDim.x + threadIdx.x;
    if (i >= total) return;
    int c = i & 127;
    float v = out[i] * scale[c] + shift[c];
    out[i] = (v > 0.f) ? v : 0.f;
}

// ---------------------------------------------------------------------------
extern "C" void kernel_launch(void* const* d_in, const int* in_sizes, int n_in,
                              void* d_out, int out_size, void* d_ws, size_t ws_size,
                              hipStream_t stream) {
    const float* x         = (const float*)d_in[0];
    const int*   eidx      = (const int*)  d_in[1];
    const float* edge_attr = (const float*)d_in[2];
    const float* Wf        = (const float*)d_in[3];
    const float* a_src_f   = (const float*)d_in[4];
    const float* a_dst_f   = (const float*)d_in[5];
    const float* We_f      = (const float*)d_in[6];
    const float* a_e_f     = (const float*)d_in[7];
    const float* bf        = (const float*)d_in[8];
    const float* Wb        = (const float*)d_in[9];
    const float* a_src_b   = (const float*)d_in[10];
    const float* a_dst_b   = (const float*)d_in[11];
    const float* We_b      = (const float*)d_in[12];
    const float* a_e_b     = (const float*)d_in[13];
    const float* bb        = (const float*)d_in[14];
    const float* Wc        = (const float*)d_in[15];
    const float* bc        = (const float*)d_in[16];
    const float* gamma     = (const float*)d_in[17];
    const float* beta      = (const float*)d_in[18];
    float* out = (float*)d_out;

    const int N = in_sizes[0] / N_IN;   // 50000
    const int E = in_sizes[1] / 2;      // 800000
    const int* src = eidx;
    const int* dst = eidx + E;

    // workspace carve (floats)
    float* ws    = (float*)d_ws;
    float* h     = ws;                      // N*128
    float* comb  = h + (size_t)N * N_OUT;   // N*128
    float* asf   = comb + (size_t)N * N_OUT;
    float* adf   = asf + N;
    float* asb   = adf + N;
    float* adb   = asb + N;
    float* m_f   = adb + N;
    float* m_b   = m_f + N;
    float* den_f = m_b + N;
    float* den_b = den_f + N;
    float* logit = den_b + N;               // E
    float* exv   = logit + E;               // E
    int*   emask = (int*)(exv + E);         // E
    float* esc   = (float*)(emask + E);     // 16
    float* scale = esc + 16;                // 128
    float* shift = scale + N_OUT;           // 128

    const int rowBlocks = (N + ROWS_PER_BLK - 1) / ROWS_PER_BLK;

    gemm_h_kernel<<<rowBlocks, 256, 0, stream>>>(x, Wf, Wb, h, N);
    escore_kernel<<<1, 16, 0, stream>>>(We_f, a_e_f, We_b, a_e_b, esc);
    node_alpha_kernel<<<(N + 255) / 256, 256, 0, stream>>>(
        h, a_src_f, a_dst_f, a_src_b, a_dst_b, asf, adf, asb, adb, N);
    init_kernel<<<(N * N_OUT + 255) / 256, 256, 0, stream>>>(
        comb, bf, bb, m_f, m_b, den_f, den_b, N);
    edge_pass1_kernel<<<(E + 255) / 256, 256, 0, stream>>>(
        src, dst, edge_attr, asf, adf, asb, adb, esc, logit, emask, m_f, m_b, E);
    edge_pass2_kernel<<<(E + 255) / 256, 256, 0, stream>>>(
        dst, logit, emask, m_f, m_b, den_f, den_b, exv, E);
    {
        long long tot = (long long)E * 64;
        int blocks = (int)((tot + 255) / 256);
        edge_scatter_kernel<<<blocks, 256, 0, stream>>>(
            src, dst, emask, exv, den_f, den_b, h, comb, E);
    }
    gemm_combine_kernel<<<rowBlocks, 256, 0, stream>>>(comb, Wc, bc, out, N);
    bn_stats_kernel<<<N_OUT, 256, 0, stream>>>(out, gamma, beta, scale, shift, N);
    bn_apply_kernel<<<(N * N_OUT + 255) / 256, 256, 0, stream>>>(
        out, scale, shift, N * N_OUT);
}
